// ResBlock_deformable3d_89653147337271
// MI455X (gfx1250) — compile-verified, matrix operand order checked by reading, not run on hardware
//
#include <hip/hip_runtime.h>
#include <hip/hip_bf16.h>

// ---------------------------------------------------------------------------
// Deformable 3D ResBlock (dimension='HW') for MI455X / gfx1250 (wave32, WMMA)
//
//   h = LeakyReLU( DeformConv3d(x; w_off0,b_off0,w0,b0) )
//   y = DeformConv3d(h; w_off1,b_off1,w1,b1) + x
//
// Both the offset conv (plain 3x3x3) and the deformable conv are implemented
// as implicit GEMMs on v_wmma_f32_16x16x32_f16: M=out-channels, K=27 taps x 48
// in-channels (padded to 64/tap), N=16 spatial positions per wave.
// ---------------------------------------------------------------------------

typedef _Float16 v16h __attribute__((ext_vector_type(16)));
typedef _Float16 v2h  __attribute__((ext_vector_type(2)));
typedef float    v8f  __attribute__((ext_vector_type(8)));

constexpr int kB   = 4;
constexpr int kC   = 48;
constexpr int kD   = 12;
constexpr int kH   = 56;
constexpr int kW   = 56;
constexpr int kHW  = kH * kW;        // 3136
constexpr int kDHW = kD * kHW;       // 37632
constexpr int kTilesPerB = kDHW / 64;  // 588 (4 waves x 16 positions / block)
constexpr float kNegSlope = 0.1f;

// K-permutation for the 16-bit WMMA operand layout (ISA 7.12.2):
//   lanes 0-15 read halves = K{0..7,16..23}; lanes 16-31 read K{8..15,24..31}.
__device__ __forceinline__ int kslot(int kl) {
  return (kl & 7) + ((kl >> 3) & 1) * 16 + ((kl >> 4) & 1) * 8;
}

// ---------------------------------------------------------------------------
// Pack fp32 OIDHW conv weights into lane-exact f16 A-fragments:
//   dst[(((tap*2+kc)*MT + mt)*32 + lane)*16 + i]
// so the GEMM loop loads one contiguous v16h (32B) per lane per fragment.
// ---------------------------------------------------------------------------
__global__ __launch_bounds__(256) void pack_weights_kernel(
    const float* __restrict__ w, _Float16* __restrict__ dst, int Mreal, int MT) {
  int idx = blockIdx.x * 256 + threadIdx.x;
  int total = 27 * 2 * MT * 512;
  if (idx >= total) return;
  int i    = idx & 15;
  int lane = (idx >> 4) & 31;
  int rest = idx >> 9;
  int mt = rest % MT;
  int kc = (rest / MT) & 1;
  int t  = rest / (MT * 2);
  int m  = lane & 15;
  int hi = lane >> 4;
  int kl = (i < 8) ? (i + hi * 8) : (i + 8 + hi * 8);  // inverse of kslot()
  int c  = kc * 32 + kl;
  int o  = mt * 16 + m;
  float v = (o < Mreal && c < kC) ? w[(o * kC + c) * 27 + t] : 0.0f;
  dst[idx] = (_Float16)v;
}

// ---------------------------------------------------------------------------
// Fused implicit-GEMM conv kernel.
//   MT=4, DEFORM=false : offset conv (48->54 ch padded to 64), out=offbuf
//   MT=3, DEFORM=true  : deformable conv (48->48), bias+ReLU/residual fused
// Block = 128 threads = 4 waves; each wave owns 16 consecutive flat positions.
// ---------------------------------------------------------------------------
template <int MT, bool DEFORM, bool RELU, bool RESID>
__global__ __launch_bounds__(128) void conv_gemm_kernel(
    const float* __restrict__ X, const float* __restrict__ offs,
    const _Float16* __restrict__ wfrag, const float* __restrict__ bias,
    const float* __restrict__ resid, float* __restrict__ out) {
  // per-wave B staging: [wave][kc=2][col=16][48 halves] ; 96B col stride keeps
  // v16h reads 32B-aligned; K=48..63 pad slots stay zero forever.
  __shared__ __align__(32) _Float16 smem[4 * 2 * 16 * 48];
  for (int i = threadIdx.x; i < 4 * 2 * 16 * 48; i += 128)
    smem[i] = (_Float16)0.0f;
  __syncthreads();

  const int lane = threadIdx.x & 31;
  const int wid  = threadIdx.x >> 5;
  const int n    = lane & 15;   // position within N-tile / B-matrix column
  const int cg   = lane >> 4;   // half-wave group

  const int tile = blockIdx.x;
  const int b    = tile / kTilesPerB;
  const int tb   = tile - b * kTilesPerB;
  const int pos  = (tb * 4 + wid) * 16 + n;     // flat index into D*H*W
  const int d  = pos / kHW;
  const int rm = pos - d * kHW;
  const int h  = rm / kW;
  const int wv = rm - h * kW;

  v8f acc[MT];
  v8f vzero = {};
#pragma unroll
  for (int mt = 0; mt < MT; ++mt) acc[mt] = vzero;

  const size_t xb = (size_t)b * kC * kDHW;
  const v16h* A = (const v16h*)wfrag;
  const int colbase = ((wid * 2) * 16 + n) * 48;  // kc==0; kc stride = 16*48

  for (int t = 0; t < 27; ++t) {
    const int kd = t / 9 - 1;
    const int kh = (t / 3) % 3 - 1;
    const int kw = t % 3 - 1;

    const int  pd  = d + kd;
    const bool vd  = (pd >= 0) & (pd < kD);
    const int  pdc = pd < 0 ? 0 : (pd >= kD ? kD - 1 : pd);
    const float* cbase = X + xb + (size_t)pdc * kHW;

    if (DEFORM) {
      if (t < 26)  // warm next tap's offsets (global_prefetch_b8)
        __builtin_prefetch(&offs[((size_t)b * 54 + (t + 1) * 2) * kDHW + pos], 0, 0);
      const float dh = offs[((size_t)b * 54 + t * 2 + 0) * kDHW + pos];
      const float dw = offs[((size_t)b * 54 + t * 2 + 1) * kDHW + pos];
      const float ph = (float)(h + kh) + dh;
      const float pw = (float)(wv + kw) + dw;
      const float h0f = floorf(ph), w0f = floorf(pw);
      const float th = ph - h0f, tw = pw - w0f;
      const int h0 = (int)h0f, w0 = (int)w0f;
      const int h1 = h0 + 1,   w1 = w0 + 1;
      const bool vh0 = (h0 >= 0) & (h0 < kH), vh1 = (h1 >= 0) & (h1 < kH);
      const bool vw0 = (w0 >= 0) & (w0 < kW), vw1 = (w1 >= 0) & (w1 < kW);
      const int h0c = h0 < 0 ? 0 : (h0 >= kH ? kH - 1 : h0);
      const int h1c = h1 < 0 ? 0 : (h1 >= kH ? kH - 1 : h1);
      const int w0c = w0 < 0 ? 0 : (w0 >= kW ? kW - 1 : w0);
      const int w1c = w1 < 0 ? 0 : (w1 >= kW ? kW - 1 : w1);
      const float g00 = (vd & vh0 & vw0) ? (1.f - th) * (1.f - tw) : 0.f;
      const float g01 = (vd & vh0 & vw1) ? (1.f - th) * tw : 0.f;
      const float g10 = (vd & vh1 & vw0) ? th * (1.f - tw) : 0.f;
      const float g11 = (vd & vh1 & vw1) ? th * tw : 0.f;
      const float* p00 = cbase + h0c * kW + w0c;
      const float* p01 = cbase + h0c * kW + w1c;
      const float* p10 = cbase + h1c * kW + w0c;
      const float* p11 = cbase + h1c * kW + w1c;
#pragma unroll 4
      for (int p = 0; p < 12; ++p) {           // each lane: 12 channel pairs
        const int c0 = cg * 24 + p * 2;
        const size_t co = (size_t)c0 * kDHW;
        float s0 = g00 * p00[co] + g01 * p01[co] + g10 * p10[co] + g11 * p11[co];
        float s1 = g00 * p00[co + kDHW] + g01 * p01[co + kDHW] +
                   g10 * p10[co + kDHW] + g11 * p11[co + kDHW];
        v2h pr = {(_Float16)s0, (_Float16)s1};
        *(v2h*)&smem[colbase + (c0 >> 5) * (16 * 48) + kslot(c0 & 31)] = pr;
      }
    } else {  // plain integer-tap conv (offset conv), zero-padded borders
      const int ph = h + kh, pw = wv + kw;
      const bool v = vd & (ph >= 0) & (ph < kH) & (pw >= 0) & (pw < kW);
      const float gv = v ? 1.f : 0.f;
      const float* p00 = cbase + (v ? (ph * kW + pw) : 0);
#pragma unroll 4
      for (int p = 0; p < 12; ++p) {
        const int c0 = cg * 24 + p * 2;
        const size_t co = (size_t)c0 * kDHW;
        v2h pr = {(_Float16)(gv * p00[co]), (_Float16)(gv * p00[co + kDHW])};
        *(v2h*)&smem[colbase + (c0 >> 5) * (16 * 48) + kslot(c0 & 31)] = pr;
      }
    }

    // LDS is processed in-order within a wave; fence the compiler + data.
    __builtin_amdgcn_wave_barrier();
    asm volatile("s_wait_dscnt 0" ::: "memory");

    const int rbase = ((wid * 2) * 16 + n) * 48 + cg * 16;
    v16h bf0 = *(const v16h*)&smem[rbase];
    v16h bf1 = *(const v16h*)&smem[rbase + 16 * 48];
#pragma unroll
    for (int mt = 0; mt < MT; ++mt) {
      v16h a0 = A[((t * 2 + 0) * MT + mt) * 32 + lane];
      v16h a1 = A[((t * 2 + 1) * MT + mt) * 32 + lane];
      acc[mt] = __builtin_amdgcn_wmma_f32_16x16x32_f16(
          false, a0, false, bf0, (short)0, acc[mt], false, false);
      acc[mt] = __builtin_amdgcn_wmma_f32_16x16x32_f16(
          false, a1, false, bf1, (short)0, acc[mt], false, false);
    }
    __builtin_amdgcn_wave_barrier();
  }

  // C/D layout: vgpr r, lane l -> (m = 8*(l>=16) + r, n = l&15)
#pragma unroll
  for (int mt = 0; mt < MT; ++mt) {
#pragma unroll
    for (int r = 0; r < 8; ++r) {
      const int o = mt * 16 + cg * 8 + r;
      float val = acc[mt][r];
      if (DEFORM) {
        val += bias[o];
        if (RELU) val = val >= 0.f ? val : kNegSlope * val;
        if (RESID) val += resid[((size_t)b * kC + o) * kDHW + pos];
        out[((size_t)b * kC + o) * kDHW + pos] = val;
      } else {
        if (o < 54) out[((size_t)b * 54 + o) * kDHW + pos] = val + bias[o];
      }
    }
  }
}

// ---------------------------------------------------------------------------
// Host launcher. Workspace layout (bytes):
//   [0)        packed f16 weight fragments (off0 | w0 | off1 | w1)   ~0.76 MB
//   [1MB)      offset buffer  [B][54][D*H*W] fp32                    ~32.5 MB
//   [+)        intermediate h [B][48][D*H*W] fp32                    ~28.9 MB
// ---------------------------------------------------------------------------
extern "C" void kernel_launch(void* const* d_in, const int* in_sizes, int n_in,
                              void* d_out, int out_size, void* d_ws, size_t ws_size,
                              hipStream_t stream) {
  (void)in_sizes; (void)n_in; (void)out_size; (void)ws_size;
  const float* x     = (const float*)d_in[0];
  const float* woff0 = (const float*)d_in[1];
  const float* boff0 = (const float*)d_in[2];
  const float* w0    = (const float*)d_in[3];
  const float* b0    = (const float*)d_in[4];
  const float* woff1 = (const float*)d_in[5];
  const float* boff1 = (const float*)d_in[6];
  const float* w1    = (const float*)d_in[7];
  const float* b1    = (const float*)d_in[8];
  float* out = (float*)d_out;

  char* ws = (char*)d_ws;
  _Float16* f_off0 = (_Float16*)(ws + 0);       // 27*2*4*512 halves = 221184 B
  _Float16* f_w0   = (_Float16*)(ws + 221184);  // 27*2*3*512 halves = 165888 B
  _Float16* f_off1 = (_Float16*)(ws + 387072);
  _Float16* f_w1   = (_Float16*)(ws + 608256);
  float* offbuf = (float*)(ws + (1 << 20));
  float* hbuf   = (float*)(ws + (1 << 20) + (size_t)kB * 54 * kDHW * sizeof(float));

  // pack weights -> f16 A-fragments
  pack_weights_kernel<<<(27 * 2 * 4 * 512 + 255) / 256, 256, 0, stream>>>(woff0, f_off0, 54, 4);
  pack_weights_kernel<<<(27 * 2 * 3 * 512 + 255) / 256, 256, 0, stream>>>(w0,    f_w0,   48, 3);
  pack_weights_kernel<<<(27 * 2 * 4 * 512 + 255) / 256, 256, 0, stream>>>(woff1, f_off1, 54, 4);
  pack_weights_kernel<<<(27 * 2 * 3 * 512 + 255) / 256, 256, 0, stream>>>(w1,    f_w1,   48, 3);

  const dim3 grid(kB * kTilesPerB);  // 2352 workgroups x 128 threads

  // layer 1: offsets, then deformable conv + bias + LeakyReLU -> hbuf
  conv_gemm_kernel<4, false, false, false><<<grid, 128, 0, stream>>>(
      x, nullptr, f_off0, boff0, nullptr, offbuf);
  conv_gemm_kernel<3, true, true, false><<<grid, 128, 0, stream>>>(
      x, offbuf, f_w0, b0, nullptr, hbuf);

  // layer 2: offsets from h, deformable conv + bias + residual(x) -> out
  conv_gemm_kernel<4, false, false, false><<<grid, 128, 0, stream>>>(
      hbuf, nullptr, f_off1, boff1, nullptr, offbuf);
  conv_gemm_kernel<3, true, false, true><<<grid, 128, 0, stream>>>(
      hbuf, offbuf, f_w1, b1, x, out);
}
